// LatentODE_32933809225792
// MI455X (gfx1250) — compile-verified, matrix operand order checked by reading, not run on hardware
//
#include <hip/hip_runtime.h>
#include <math.h>

typedef __attribute__((ext_vector_type(16))) _Float16 v16h;
typedef __attribute__((ext_vector_type(8)))  float    v8f;
typedef int b128_t __attribute__((vector_size(16)));   // int4, matches builtin signature

namespace {
constexpr int T_STEPS = 256;
constexpr int BATCH   = 512;
constexpr int DDIM    = 128;
constexpr int SDIM    = 256;   // latent dim (== R)
constexpr int HDIM    = 512;
constexpr int CHDIM   = 64;
constexpr int NSAMP   = 8;
constexpr int BTILE   = 64;                 // batch rows per workgroup
constexpr int NWG     = BATCH / BTILE;      // 8 workgroups
constexpr int ZSTRIDE = SDIM + 1;           // padded f32 row (bank-conflict free)
constexpr int ASTRIDE = HDIM + 4;           // padded f16 row
constexpr float LOG2PI_F  = 1.8378770664093453f;
constexpr float DT_FACTOR = 0.005f;         // 0.1 * dt

// workspace layout (bytes)
constexpr size_t WS_KLD_OFF = 0;                        // 512 floats (per-block partials)
constexpr size_t WS_REC_OFF = 2048;                     // 8 floats
constexpr size_t WS_Z0_OFF  = 4096;                     // 512*256 f32
constexpr size_t WS_PW_OFF  = WS_Z0_OFF + (size_t)BATCH*SDIM*4;  // packed f16 weights
constexpr int    PW_ELEMS   = 131072;                   // halves per weight (K*N)

// dynamic LDS carve (bytes)
constexpr size_t L_ZS   = 0;                                   // BTILE*ZSTRIDE f32
constexpr size_t L_A1   = L_ZS  + (size_t)BTILE * ZSTRIDE * 4; // BTILE*ASTRIDE f16
constexpr size_t L_XB   = L_A1  + (size_t)BTILE * ASTRIDE * 2; // 2 * BTILE*DDIM f32
constexpr size_t L_MB   = L_XB  + (size_t)2 * BTILE * DDIM * 4;
constexpr size_t L_BO1  = L_MB  + (size_t)2 * BTILE * DDIM * 4; // HDIM f32
constexpr size_t L_BO2  = L_BO1 + (size_t)HDIM * 4;             // SDIM f32
constexpr size_t L_BP1  = L_BO2 + (size_t)SDIM * 4;             // HDIM f32
constexpr size_t L_BP2  = L_BP1 + (size_t)HDIM * 4;             // SDIM f32
constexpr size_t L_MT   = L_BP2 + (size_t)SDIM * 4;             // BTILE f32
constexpr size_t L_RED  = L_MT  + (size_t)BTILE * 4;            // 256 f32
constexpr size_t L_TOTAL = L_RED + 256 * 4;                     // 270,336 B < 320 KB
}

// ---------------------------------------------------------------------------
// CDNA5 async global->LDS copy (ASYNCcnt-tracked), 16B per lane.
// ---------------------------------------------------------------------------
__device__ __forceinline__ void async_copy_b128(void* lds_dst, const void* gsrc) {
#if __has_builtin(__builtin_amdgcn_global_load_async_to_lds_b128)
  __builtin_amdgcn_global_load_async_to_lds_b128(
      (__attribute__((address_space(1))) b128_t*)(uintptr_t)gsrc,
      (__attribute__((address_space(3))) b128_t*)(unsigned)(uintptr_t)lds_dst,
      0, 0);
#else
  const unsigned lds_off = (unsigned)(uintptr_t)lds_dst;  // low 32 bits = LDS byte addr
  asm volatile("global_load_async_to_lds_b128 %0, %1, off"
               :: "v"(lds_off), "v"(gsrc) : "memory");
#endif
}

__device__ __forceinline__ void wait_async_zero() {
#if __has_builtin(__builtin_amdgcn_s_wait_asynccnt)
  __builtin_amdgcn_s_wait_asynccnt(0);
#else
  asm volatile("s_wait_asynccnt 0x0" ::: "memory");
#endif
}

// K index inside a 32-deep f16 fragment for (half i, lane) per ISA 7.12.2
__device__ __forceinline__ int kmap(int i, int lane) {
  return (i & 7) + ((i & 8) << 1) + ((lane & 16) ? 8 : 0);
}

__device__ __forceinline__ v8f wmma_f16(v16h a, v16h b, v8f c) {
  return __builtin_amdgcn_wmma_f32_16x16x32_f16(false, a, false, b, (short)0, c,
                                                false, false);
}

// ---------------------------------------------------------------------------
// Repack W (K x N, row-major f32) into f16 WMMA B-fragment order:
// packed[((nt*KO + ko)*32 + lane)*16 + i] = W[(ko*32 + kmap(i,lane)) * N + nt*16 + (lane&15)]
// ---------------------------------------------------------------------------
__global__ void repack_weights(const float* __restrict__ Wo1, const float* __restrict__ Wo2,
                               const float* __restrict__ Wp1, const float* __restrict__ Wp2,
                               _Float16* __restrict__ pWo1, _Float16* __restrict__ pWo2,
                               _Float16* __restrict__ pWp1, _Float16* __restrict__ pWp2) {
  const int total = 4 * PW_ELEMS;
  for (int p = blockIdx.x * blockDim.x + threadIdx.x; p < total;
       p += gridDim.x * blockDim.x) {
    const int w = p >> 17;
    const int q = p & (PW_ELEMS - 1);
    const float* src; _Float16* dst; int K, N;
    if      (w == 0) { src = Wo1; dst = pWo1; K = SDIM; N = HDIM; }
    else if (w == 1) { src = Wo2; dst = pWo2; K = HDIM; N = SDIM; }
    else if (w == 2) { src = Wp1; dst = pWp1; K = SDIM; N = HDIM; }
    else             { src = Wp2; dst = pWp2; K = HDIM; N = SDIM; }
    const int i    = q & 15;
    const int lane = (q >> 4) & 31;
    const int rest = q >> 9;
    const int KO   = K >> 5;
    const int ko   = rest % KO;
    const int nt   = rest / KO;
    const int n    = nt * 16 + (lane & 15);
    const int k    = ko * 32 + kmap(i, lane);
    dst[q] = (_Float16)src[k * N + n];
  }
}

// ---------------------------------------------------------------------------
// Encoder: z0 = mu + exp(0.5*logvar)*eps ; per-block KLD partial
// ---------------------------------------------------------------------------
__global__ __launch_bounds__(128)
void init_encoder(const float* __restrict__ h0,  const float* __restrict__ eps,
                  const float* __restrict__ Wi1, const float* __restrict__ bi1,
                  const float* __restrict__ Wi2, const float* __restrict__ bi2,
                  float* __restrict__ z0, float* __restrict__ kld_part) {
  __shared__ float hid[CHDIM];
  __shared__ float out2[2 * SDIM];
  __shared__ float red[128];
  const int b = blockIdx.x, tid = threadIdx.x;
  if (tid < CHDIM) {
    float s = 0.f;
    for (int k = 0; k < SDIM; ++k) s += h0[b * SDIM + k] * Wi1[k * CHDIM + tid];
    s += bi1[tid];
    hid[tid] = s > 0.f ? s : 0.f;
  }
  __syncthreads();
  for (int j = tid; j < 2 * SDIM; j += 128) {
    float s = 0.f;
    for (int k = 0; k < CHDIM; ++k) s += hid[k] * Wi2[k * 2 * SDIM + j];
    out2[j] = tanhf(s + bi2[j]);
  }
  __syncthreads();
  float kacc = 0.f;
  for (int s = tid; s < SDIM; s += 128) {
    const float mu = out2[s], lv = out2[SDIM + s];
    z0[b * SDIM + s] = mu + __expf(0.5f * lv) * eps[b * SDIM + s];
    kacc += 0.5f * (__expf(lv) + mu * mu - 1.f - lv);
  }
  red[tid] = kacc;
  __syncthreads();
  if (tid == 0) {
    float s = 0.f;
    for (int i = 0; i < 128; ++i) s += red[i];
    kld_part[b] = s * (1.0f / ((float)BATCH * SDIM));
  }
}

// ---------------------------------------------------------------------------
// dense (BTILE x HDIM) = relu( z(f32,LDS) @ W(S x H) + bias(LDS) ) -> a1 (f16, LDS)
// ---------------------------------------------------------------------------
__device__ __forceinline__ void dense_relu_from_z(
    const float* __restrict__ zs, _Float16* __restrict__ a1,
    const v16h* __restrict__ pW, const float* __restrict__ bias,
    int lane, int wave) {
  const int khi = (lane & 16) ? 8 : 0;
  for (int p = wave; p < (BTILE / 16) * (HDIM / 16); p += 8) {
    const int mt = p & 3;
    const int nt = p >> 2;
    v8f acc = {};
    const float* zr = zs + (mt * 16 + (lane & 15)) * ZSTRIDE;
    for (int ko = 0; ko < SDIM / 32; ++ko) {
      v16h a;
      const int kb = ko * 32 + khi;
#pragma unroll
      for (int i = 0; i < 16; ++i)
        a[i] = (_Float16)zr[kb + (i & 7) + ((i & 8) << 1)];
      const v16h bfrag = pW[(nt * (SDIM / 32) + ko) * 32 + lane];
      acc = wmma_f16(a, bfrag, acc);
    }
    const int n  = nt * 16 + (lane & 15);
    const float bv = bias[n];
    const int mb = mt * 16 + ((lane >> 4) & 1) * 8;
#pragma unroll
    for (int r = 0; r < 8; ++r) {
      const float v = acc[r] + bv;
      a1[(mb + r) * ASTRIDE + n] = (_Float16)(v > 0.f ? v : 0.f);
    }
  }
}

__device__ __forceinline__ v16h load_afrag_f16(const _Float16* __restrict__ ar, int kb) {
  v16h a;
#pragma unroll
  for (int i = 0; i < 16; ++i)
    a[i] = ar[kb + (i & 7) + ((i & 8) << 1)];
  return a;
}

// ---------------------------------------------------------------------------
// Persistent scan: one workgroup owns BTILE batch rows for all 256 steps.
// X/M tiles double-buffered in LDS via async global->LDS copies.
// ---------------------------------------------------------------------------
__global__ __launch_bounds__(256)
void latent_ode_scan(const float* __restrict__ X, const float* __restrict__ M,
                     const float* __restrict__ z0,
                     const float* __restrict__ bo1, const float* __restrict__ bo2,
                     const float* __restrict__ bp1, const float* __restrict__ bp2,
                     const v16h* __restrict__ pWo1, const v16h* __restrict__ pWo2,
                     const v16h* __restrict__ pWp1, const v16h* __restrict__ pWp2,
                     float* __restrict__ zout, float* __restrict__ rec_part) {
  extern __shared__ char smem[];
  float*    zs   = (float*)(smem + L_ZS);
  _Float16* a1   = (_Float16*)(smem + L_A1);
  float*    Xb   = (float*)(smem + L_XB);   // 2 tiles of BTILE*DDIM
  float*    Mb   = (float*)(smem + L_MB);   // 2 tiles of BTILE*DDIM
  float*    bo1s = (float*)(smem + L_BO1);
  float*    bo2s = (float*)(smem + L_BO2);
  float*    bp1s = (float*)(smem + L_BP1);
  float*    bp2s = (float*)(smem + L_BP2);
  float*    Mt   = (float*)(smem + L_MT);
  float*    red  = (float*)(smem + L_RED);

  const int tid  = threadIdx.x;
  const int lane = tid & 31;
  const int wave = tid >> 5;
  const int b0   = blockIdx.x * BTILE;
  const int khi  = (lane & 16) ? 8 : 0;
  constexpr int TILE_F32 = BTILE * DDIM;      // 8192 floats = 32 KB
  constexpr int COPIES   = TILE_F32 / (4 * 256);  // 8 b128 copies per thread

  // preload z0 and biases
  for (int idx = tid; idx < BTILE * SDIM; idx += 256) {
    const int m = idx >> 8, s = idx & 255;
    zs[m * ZSTRIDE + s] = z0[(b0 + m) * SDIM + s];
  }
  for (int i = tid; i < HDIM; i += 256) { bo1s[i] = bo1[i]; bp1s[i] = bp1[i]; }
  for (int i = tid; i < SDIM; i += 256) { bo2s[i] = bo2[i]; bp2s[i] = bp2[i]; }

  // prologue: async-stage X[0], M[0] into buffer 0
  {
    const float* Xn = X + (size_t)b0 * DDIM;
    const float* Mn = M + (size_t)b0 * DDIM;
#pragma unroll
    for (int j = 0; j < COPIES; ++j) {
      const int idx = (tid + j * 256) * 4;
      async_copy_b128(Xb + idx, Xn + idx);
      async_copy_b128(Mb + idx, Mn + idx);
    }
  }
  float recon = 0.f;
  wait_async_zero();
  __syncthreads();

  for (int t = 0; t < T_STEPS; ++t) {
    const int buf = t & 1;
    const float* Xt  = Xb + buf * TILE_F32;
    const float* Mtl = Mb + buf * TILE_F32;

    // kick off async staging of next step's tiles into the other buffer
    if (t + 1 < T_STEPS) {
      const float* Xn = X + (size_t)(t + 1) * BATCH * DDIM + (size_t)b0 * DDIM;
      const float* Mn = M + (size_t)(t + 1) * BATCH * DDIM + (size_t)b0 * DDIM;
      float* Xd = Xb + (buf ^ 1) * TILE_F32;
      float* Md = Mb + (buf ^ 1) * TILE_F32;
#pragma unroll
      for (int j = 0; j < COPIES; ++j) {
        const int idx = (tid + j * 256) * 4;
        async_copy_b128(Xd + idx, Xn + idx);
        async_copy_b128(Md + idx, Mn + idx);
      }
    }

    // mask row means from LDS tile (cooperative: 4 threads per row)
    {
      const int row = tid >> 2, q = tid & 3;
      const float* r = Mtl + row * DDIM + q * 32;
      float s = 0.f;
#pragma unroll 8
      for (int j = 0; j < 32; ++j) s += r[j];
      red[tid] = s;
    }
    __syncthreads();
    if (tid < BTILE)
      Mt[tid] = (red[tid * 4] + red[tid * 4 + 1] + red[tid * 4 + 2] + red[tid * 4 + 3]) *
                (1.0f / DDIM);
    __syncthreads();

    // Stage A: a1 = relu(z @ Wo1 + bo1)
    dense_relu_from_z(zs, a1, pWo1, bo1s, lane, wave);
    __syncthreads();

    // Stage B: z += 0.005 * tanh(a1 @ Wo2 + bo2)
    for (int p = wave; p < (BTILE / 16) * (SDIM / 16); p += 8) {
      const int mt = p & 3;
      const int nt = p >> 2;
      v8f acc = {};
      const _Float16* ar = a1 + (mt * 16 + (lane & 15)) * ASTRIDE;
      for (int ko = 0; ko < HDIM / 32; ++ko) {
        const v16h a = load_afrag_f16(ar, ko * 32 + khi);
        const v16h bfrag = pWo2[(nt * (HDIM / 32) + ko) * 32 + lane];
        acc = wmma_f16(a, bfrag, acc);
      }
      const int n  = nt * 16 + (lane & 15);
      const float bv = bo2s[n];
      const int mb = mt * 16 + ((lane >> 4) & 1) * 8;
#pragma unroll
      for (int r = 0; r < 8; ++r)
        zs[(mb + r) * ZSTRIDE + n] += DT_FACTOR * tanhf(acc[r] + bv);
    }
    __syncthreads();

    // Stage C: a1 = relu(z @ Wp1 + bp1)
    dense_relu_from_z(zs, a1, pWp1, bp1s, lane, wave);
    __syncthreads();

    // Stage D: mean/logvar = a1 @ Wp2 + bp2 ; fused masked NLL (X from LDS)
    for (int p = wave; p < 32; p += 8) {
      const int mt  = p & 3;
      const int ntm = p >> 2;  // 0..7 -> mean tile; +8 -> logvar tile
      v8f am = {}, al = {};
      const _Float16* ar = a1 + (mt * 16 + (lane & 15)) * ASTRIDE;
      for (int ko = 0; ko < HDIM / 32; ++ko) {
        const v16h a = load_afrag_f16(ar, ko * 32 + khi);
        const v16h bm = pWp2[(ntm * (HDIM / 32) + ko) * 32 + lane];
        const v16h bl = pWp2[((ntm + 8) * (HDIM / 32) + ko) * 32 + lane];
        am = wmma_f16(a, bm, am);
        al = wmma_f16(a, bl, al);
      }
      const int nm = ntm * 16 + (lane & 15);          // 0..127 (= d index)
      const float biasm = bp2s[nm];
      const float biasl = bp2s[nm + DDIM];
      const int mb = mt * 16 + ((lane >> 4) & 1) * 8;
#pragma unroll
      for (int r = 0; r < 8; ++r) {
        const int ml = mb + r;
        const float mean = am[r] + biasm;
        const float lv   = al[r] + biasl;
        const float x    = Xt[ml * DDIM + nm];
        const float e    = x - mean;
        const float nll  = 0.5f * (LOG2PI_F + lv + e * e * __expf(-lv));
        recon += nll * Mt[ml];
      }
    }

    // next-step tiles must have landed before anyone reads them
    wait_async_zero();
    __syncthreads();
  }

  // write final z, replicated over the (collapsed) sample axis
  for (int idx = tid; idx < BTILE * SDIM; idx += 256) {
    const int m = idx >> 8, s = idx & 255;
    const float v = zs[m * ZSTRIDE + s];
    const size_t base = ((size_t)(b0 + m) * NSAMP) * SDIM + s;
#pragma unroll
    for (int rep = 0; rep < NSAMP; ++rep)
      zout[base + (size_t)rep * SDIM] = v;
  }

  red[tid] = recon;
  __syncthreads();
  if (tid == 0) {
    float s = 0.f;
    for (int i = 0; i < 256; ++i) s += red[i];
    rec_part[blockIdx.x] = s * (1.0f / BATCH);
  }
}

__global__ void finalize_losses(const float* __restrict__ kld_part,
                                const float* __restrict__ rec_part,
                                float* __restrict__ out_tail) {
  float kld = 0.f, rec = 0.f;
  for (int i = 0; i < BATCH; ++i) kld += kld_part[i];
  for (int i = 0; i < NWG; ++i)  rec += rec_part[i];
  out_tail[0] = rec + kld;
  out_tail[1] = rec;
  out_tail[2] = kld;
}

extern "C" void kernel_launch(void* const* d_in, const int* in_sizes, int n_in,
                              void* d_out, int out_size, void* d_ws, size_t ws_size,
                              hipStream_t stream) {
  const float* X   = (const float*)d_in[0];
  const float* M   = (const float*)d_in[1];
  const float* h0  = (const float*)d_in[2];
  const float* eps = (const float*)d_in[3];
  const float* Wi1 = (const float*)d_in[4];
  const float* bi1 = (const float*)d_in[5];
  const float* Wi2 = (const float*)d_in[6];
  const float* bi2 = (const float*)d_in[7];
  const float* Wp1 = (const float*)d_in[8];
  const float* bp1 = (const float*)d_in[9];
  const float* Wp2 = (const float*)d_in[10];
  const float* bp2 = (const float*)d_in[11];
  const float* Wo1 = (const float*)d_in[12];
  const float* bo1 = (const float*)d_in[13];
  const float* Wo2 = (const float*)d_in[14];
  const float* bo2 = (const float*)d_in[15];

  char* ws = (char*)d_ws;
  float*    kld_part = (float*)(ws + WS_KLD_OFF);
  float*    rec_part = (float*)(ws + WS_REC_OFF);
  float*    z0       = (float*)(ws + WS_Z0_OFF);
  _Float16* pWo1     = (_Float16*)(ws + WS_PW_OFF);
  _Float16* pWo2     = pWo1 + PW_ELEMS;
  _Float16* pWp1     = pWo2 + PW_ELEMS;
  _Float16* pWp2     = pWp1 + PW_ELEMS;

  repack_weights<<<512, 256, 0, stream>>>(Wo1, Wo2, Wp1, Wp2, pWo1, pWo2, pWp1, pWp2);
  init_encoder<<<BATCH, 128, 0, stream>>>(h0, eps, Wi1, bi1, Wi2, bi2, z0, kld_part);

  latent_ode_scan<<<NWG, 256, L_TOTAL, stream>>>(
      X, M, z0, bo1, bo2, bp1, bp2,
      (const v16h*)pWo1, (const v16h*)pWo2, (const v16h*)pWp1, (const v16h*)pWp2,
      (float*)d_out, rec_part);

  finalize_losses<<<1, 1, 0, stream>>>(
      kld_part, rec_part, (float*)d_out + (size_t)BATCH * NSAMP * SDIM);
}
